// PhyRNN_29592324669801
// MI455X (gfx1250) — compile-verified, matrix operand order checked by reading, not run on hardware
//
#include <hip/hip_runtime.h>

#define T_STEPS 3000
#define BATCH   64
#define HID     128
#define ROWS    16          // batch rows handled per block
#define DT_INV  50.0f       // 1/0.02

typedef __attribute__((ext_vector_type(16))) __bf16       v16bf;
typedef __attribute__((ext_vector_type(8)))  float        v8f;
typedef __attribute__((ext_vector_type(8)))  unsigned int v8u;

__device__ __forceinline__ unsigned pack2bf(float a, float b) {
    unsigned ua = __float_as_uint(a), ub = __float_as_uint(b);
    unsigned ra = (ua + 0x7FFFu + ((ua >> 16) & 1u)) >> 16;   // RNE f32->bf16
    unsigned rb = (ub + 0x7FFFu + ((ub >> 16) & 1u)) >> 16;
    return (rb << 16) | (ra & 0xFFFFu);
}
__device__ __forceinline__ unsigned short f2bf(float a) {
    unsigned ua = __float_as_uint(a);
    return (unsigned short)((ua + 0x7FFFu + ((ua >> 16) & 1u)) >> 16);
}
__device__ __forceinline__ float sigf(float x) { return 1.0f / (1.0f + __expf(-x)); }

__device__ __forceinline__ v8f wmma_bf16(v16bf a, v16bf b, v8f c) {
    // v_wmma_f32_16x16x32_bf16: D = A(16x32) * B(32x16) + C(16x16 f32)
    return __builtin_amdgcn_wmma_f32_16x16x32_bf16(false, a, false, b, (short)0, c,
                                                   false, false);
}

// Fragment load from an LDS row-major bf16-pair image (dwords). For 16-bit WMMA
// operands (ISA 7.12.2): lane slot dwords live at rowBase + k0/2 + 4*hi +
// {0,1,2,3, 8,9,10,11} -> exactly two ds_load_b128, no divergence.
// Works identically for A tiles (row = m) and W-row-major B tiles (row = n),
// since gates[m][n] = sum_k A[m][k] * W[n][k].
__device__ __forceinline__ v16bf load_frag(const unsigned int* base, int row,
                                           int rowDw, int k0, int hi) {
    v8u u;
    const unsigned int* r = base + row * rowDw + (k0 >> 1) + 4 * hi;
#pragma unroll
    for (int v = 0; v < 8; ++v) u[v] = r[(v < 4) ? v : v + 4];
    return __builtin_bit_cast(v16bf, u);
}

// B fragment (16 rows of row-major global f32 W[n][K]) packed to bf16 registers.
__device__ __forceinline__ v16bf load_bfrag_g(const float* __restrict__ W, int n, int K,
                                              int k0, int kbase) {
    v8u u;
    const float* r = W + n * K;
#pragma unroll
    for (int v = 0; v < 8; ++v) {
        int kk = k0 + kbase + ((v < 4) ? 2 * v : 16 + 2 * (v - 4));
        u[v] = pack2bf(r[kk], r[kk + 1]);
    }
    return __builtin_bit_cast(v16bf, u);
}

// Fused: 2-layer LSTM scan + MLP head. One block per 16-batch slice (grid=4),
// 256 threads = 8 waves; wave w owns gate columns {g*128 + w*16 .. +16} for all
// four gates g, so i/f/g/o for one h-column slice stay inside one wave and the
// cell state c lives in registers across all 3000 steps. All three K=128 weight
// matrices are register-resident bf16 WMMA fragments (loaded once). Layer-0's
// K=8 input projection (zero-padded to K=32) and the 16x16 fc1 head also run
// on WMMA, with physically zero-padded LDS images so fragment loads stay
// branchless (2x ds_load_b128 each).
__global__ __launch_bounds__(256) void lstm_fused_scan(
    const float* __restrict__ x,                                    // [B][T][8]
    const float* __restrict__ w_ih0, const float* __restrict__ w_hh0,
    const float* __restrict__ b_ih0, const float* __restrict__ b_hh0,
    const float* __restrict__ w_ih1, const float* __restrict__ w_hh1,
    const float* __restrict__ b_ih1, const float* __restrict__ b_hh1,
    const float* __restrict__ fc1_w, const float* __restrict__ fc1_b,
    const float* __restrict__ fc2_w, const float* __restrict__ fc2_b,
    const float* __restrict__ fc3_w, const float* __restrict__ fc3_b,
    float* __restrict__ yt)                                         // [T][64]
{
    __shared__ unsigned int h0buf[2][ROWS][HID / 2];  // double-buffered bf16 h, layer0
    __shared__ unsigned int h1buf[2][ROWS][HID / 2];  // layer1
    __shared__ unsigned int wih0pad[512][16];         // w_ih0 bf16 pairs, K=8 -> 32 pad
    __shared__ unsigned int fc1wp[16][64];            // fc1_w bf16 pairs (K=128)
    __shared__ unsigned int xpk[ROWS][16];            // x_t bf16 pairs, K=8 -> 32 pad
    __shared__ float fc1b_l[16], fc2w_l[4][16], fc2b_l[4], fc3w_l[4], fc3b_l[1];
    __shared__ float fc1o[ROWS][16];

    const int tid  = threadIdx.x;
    const int blk  = blockIdx.x;          // batch slice
    const int w    = tid >> 5;            // wave 0..7
    const int lane = tid & 31;
    const int hi   = lane >> 4;           // half-wave
    const int nl   = lane & 15;
    const int kbase = 8 * hi;

    // ---- one-time staging ----
    for (int i = tid; i < 512 * 16; i += 256) {
        int n = i >> 4, d = i & 15;
        unsigned val = 0u;
        if (d < 4) val = pack2bf(w_ih0[n * 8 + 2 * d], w_ih0[n * 8 + 2 * d + 1]);
        wih0pad[n][d] = val;               // dwords 4..15 are the K padding
    }
    for (int i = tid; i < 16 * 64; i += 256) {
        int j = i >> 6, d = i & 63;
        fc1wp[j][d] = pack2bf(fc1_w[j * HID + 2 * d], fc1_w[j * HID + 2 * d + 1]);
    }
    if (tid < 16) fc1b_l[tid] = fc1_b[tid];
    if (tid < 64) fc2w_l[tid >> 4][tid & 15] = fc2_w[tid];
    if (tid < 4) { fc2b_l[tid] = fc2_b[tid]; fc3w_l[tid] = fc3_w[tid]; }
    if (tid == 0) fc3b_l[0] = fc3_b[0];
    for (int i = tid; i < ROWS * (HID / 2); i += 256) {
        h0buf[0][i >> 6][i & 63] = 0u;     // h(-1) = 0
        h1buf[0][i >> 6][i & 63] = 0u;
    }

    // ---- register-resident weight fragments (loaded once, reused 3000x) ----
    v16bf fh0[4][4], fi1[4][4], fh1[4][4];   // [gate][k-tile]
    float bs0[4], bs1[4];
#pragma unroll
    for (int gt = 0; gt < 4; ++gt) {
        const int n = (gt * 8 + w) * 16 + nl;        // gate row in [0,512)
        bs0[gt] = b_ih0[n] + b_hh0[n];
        bs1[gt] = b_ih1[n] + b_hh1[n];
#pragma unroll
        for (int k = 0; k < 4; ++k) {
            fh0[gt][k] = load_bfrag_g(w_hh0, n, HID, k * 32, kbase);
            fi1[gt][k] = load_bfrag_g(w_ih1, n, HID, k * 32, kbase);
            fh1[gt][k] = load_bfrag_g(w_hh1, n, HID, k * 32, kbase);
        }
    }

    v8f c0, c1;
#pragma unroll
    for (int e = 0; e < 8; ++e) { c0[e] = 0.0f; c1[e] = 0.0f; }

    __syncthreads();

    const int hcol = w * 16 + nl;         // this lane's h column

    for (int t = 0; t < T_STEPS; ++t) {
        const int rp = t & 1, wb = rp ^ 1;

        // stage x_t pre-packed+padded: one dword per thread
        {
            int m = tid >> 4, d = tid & 15;
            unsigned val = 0u;
            if (d < 4) {
                const float* xr = x + ((size_t)(blk * ROWS + m) * T_STEPS + t) * 8;
                val = pack2bf(xr[2 * d], xr[2 * d + 1]);
            }
            xpk[m][d] = val;
        }
        __syncthreads();

        // ---------------- layer 0 ----------------
        v8f acc[4];
#pragma unroll
        for (int gt = 0; gt < 4; ++gt)
#pragma unroll
            for (int e = 0; e < 8; ++e) acc[gt][e] = bs0[gt];
        {                                            // input proj, K=8 pad 32 (WMMA)
            v16bf ax = load_frag(&xpk[0][0], nl, 16, 0, hi);
#pragma unroll
            for (int gt = 0; gt < 4; ++gt) {
                v16bf bw = load_frag(&wih0pad[0][0], (gt * 8 + w) * 16 + nl, 16, 0, hi);
                acc[gt] = wmma_bf16(ax, bw, acc[gt]);
            }
        }
#pragma unroll
        for (int k = 0; k < 4; ++k) {                // recurrent GEMM, K=128 (WMMA)
            v16bf a = load_frag(&h0buf[rp][0][0], nl, 64, k * 32, hi);
#pragma unroll
            for (int gt = 0; gt < 4; ++gt) acc[gt] = wmma_bf16(a, fh0[gt][k], acc[gt]);
        }
#pragma unroll
        for (int e = 0; e < 8; ++e) {                // gate nonlinearity
            float c = sigf(acc[1][e]) * c0[e] + sigf(acc[0][e]) * tanhf(acc[2][e]);
            c0[e] = c;
            ((unsigned short*)&h0buf[wb][0][0])[(e + 8 * hi) * HID + hcol] =
                f2bf(sigf(acc[3][e]) * tanhf(c));
        }
        __syncthreads();

        // ---------------- layer 1 ----------------
#pragma unroll
        for (int gt = 0; gt < 4; ++gt)
#pragma unroll
            for (int e = 0; e < 8; ++e) acc[gt][e] = bs1[gt];
#pragma unroll
        for (int k = 0; k < 4; ++k) {
            v16bf a0 = load_frag(&h0buf[wb][0][0], nl, 64, k * 32, hi);  // input h0(t)
            v16bf a1 = load_frag(&h1buf[rp][0][0], nl, 64, k * 32, hi);  // h1(t-1)
#pragma unroll
            for (int gt = 0; gt < 4; ++gt) {
                acc[gt] = wmma_bf16(a0, fi1[gt][k], acc[gt]);
                acc[gt] = wmma_bf16(a1, fh1[gt][k], acc[gt]);
            }
        }
#pragma unroll
        for (int e = 0; e < 8; ++e) {
            float c = sigf(acc[1][e]) * c1[e] + sigf(acc[0][e]) * tanhf(acc[2][e]);
            c1[e] = c;
            ((unsigned short*)&h1buf[wb][0][0])[(e + 8 * hi) * HID + hcol] =
                f2bf(sigf(acc[3][e]) * tanhf(c));
        }
        __syncthreads();

        // ---------------- fused MLP head ----------------
        // fc1 is exactly one 16x16 output tile with K=128: 4 WMMAs on wave 0.
        if (w == 0) {
            v8f f1;
            const float fb = fc1b_l[nl];
#pragma unroll
            for (int e = 0; e < 8; ++e) f1[e] = fb;
#pragma unroll
            for (int k = 0; k < 4; ++k) {
                v16bf a  = load_frag(&h1buf[wb][0][0], nl, 64, k * 32, hi);
                v16bf bw = load_frag(&fc1wp[0][0], nl, 64, k * 32, hi);
                f1 = wmma_bf16(a, bw, f1);
            }
#pragma unroll
            for (int e = 0; e < 8; ++e)
                fc1o[e + 8 * hi][nl] = fmaxf(f1[e], 0.0f);   // (b, j) = (e+8*hi, nl)
        }
        __syncthreads();
        if (tid < ROWS) {                            // fc2 -> relu -> fc3
            float y = fc3b_l[0];
#pragma unroll
            for (int q = 0; q < 4; ++q) {
                float s = fc2b_l[q];
#pragma unroll
                for (int r = 0; r < 16; ++r) s += fc1o[tid][r] * fc2w_l[q][r];
                y += fmaxf(s, 0.0f) * fc3w_l[q];
            }
            yt[t * BATCH + blk * ROWS + tid] = y;    // [T][B] layout for the stencil
        }
        __syncthreads();
    }
}

// phi is the fixed central-difference operator: apply it as a 3-point stencil
// instead of a dense 3000x3000 matmul. in is [T][64]; out is [T][64] (toBT=0)
// or transposed [B][T] (toBT=1, final output layout).
__global__ __launch_bounds__(256) void phi_stencil(const float* __restrict__ in,
                                                   float* __restrict__ out, int toBT) {
    int idx = blockIdx.x * 256 + threadIdx.x;
    if (idx >= T_STEPS * BATCH) return;
    int t = idx >> 6, b = idx & 63;
    float v;
    if (t == 0)
        v = (-1.5f * in[0 * BATCH + b] + 2.0f * in[1 * BATCH + b]
             - 0.5f * in[2 * BATCH + b]) * DT_INV;
    else if (t == T_STEPS - 1)
        v = (0.5f * in[(T_STEPS - 3) * BATCH + b] - 2.0f * in[(T_STEPS - 2) * BATCH + b]
             + 1.5f * in[(T_STEPS - 1) * BATCH + b]) * DT_INV;
    else
        v = 0.5f * (in[(t + 1) * BATCH + b] - in[(t - 1) * BATCH + b]) * DT_INV;
    out[toBT ? (b * T_STEPS + t) : idx] = v;
}

extern "C" void kernel_launch(void* const* d_in, const int* in_sizes, int n_in,
                              void* d_out, int out_size, void* d_ws, size_t ws_size,
                              hipStream_t stream) {
    const float* x     = (const float*)d_in[0];
    const float* w_ih0 = (const float*)d_in[1];
    const float* w_hh0 = (const float*)d_in[2];
    const float* b_ih0 = (const float*)d_in[3];
    const float* b_hh0 = (const float*)d_in[4];
    const float* w_ih1 = (const float*)d_in[5];
    const float* w_hh1 = (const float*)d_in[6];
    const float* b_ih1 = (const float*)d_in[7];
    const float* b_hh1 = (const float*)d_in[8];
    const float* fc1_w = (const float*)d_in[9];
    const float* fc1_b = (const float*)d_in[10];
    const float* fc2_w = (const float*)d_in[11];
    const float* fc2_b = (const float*)d_in[12];
    const float* fc3_w = (const float*)d_in[13];
    const float* fc3_b = (const float*)d_in[14];

    float* yt = (float*)d_ws;                    // [T][64]
    float* z  = yt + (size_t)T_STEPS * BATCH;    // [T][64]

    lstm_fused_scan<<<BATCH / ROWS, 256, 0, stream>>>(
        x, w_ih0, w_hh0, b_ih0, b_hh0, w_ih1, w_hh1, b_ih1, b_hh1,
        fc1_w, fc1_b, fc2_w, fc2_b, fc3_w, fc3_b, yt);

    const int n = T_STEPS * BATCH;
    phi_stencil<<<(n + 255) / 256, 256, 0, stream>>>(yt, z, 0);
    phi_stencil<<<(n + 255) / 256, 256, 0, stream>>>(z, (float*)d_out, 1);
}
